// DimeNet_48842368090498
// MI455X (gfx1250) — matrix operand order repaired and checked by reading, not compiled, not run on hardware
//
#include <hip/hip_runtime.h>
#include <hip/hip_bf16.h>
#include <math.h>

// ---------------------------------------------------------------------------
// DimeNet++ forward for MI455X (gfx1250, wave32).
// Dense 128/256-wide GEMMs via V_WMMA_F32_16X16X32_BF16 (bf16 A/B, f32 acc).
// Each wave register-blocks 4 M-tiles (64 rows) against one shared B fragment
// so the weight loads / bf16 converts are amortized 4x and WMMA density per
// inner loop is 4. A tiles staged through LDS; tiny-K basis layers on VALU;
// segment sums via index contiguity (edge_i) or global_atomic_add_f32.
// ---------------------------------------------------------------------------

#define HIDDEN   128
#define OUT_EMB  256
#define INT_EMB  64
#define NUM_SPH  7
#define NUM_RAD  6
#define DEG      6
#define CUTOFF_F 5.0f
#define MT       4        // M-tiles per wave (64 rows per block)

typedef __attribute__((ext_vector_type(16))) __bf16 v16bf;
typedef __attribute__((ext_vector_type(8)))  float  v8f;

__device__ __forceinline__ float siluf(float v) {
    return v / (1.0f + __expf(-v));
}

// ---------------------------------------------------------------------------
// Generic WMMA GEMM: C[M,N] = epi(A[M,K] @ W[K,N])
// epi order: +bias -> silu(if act) -> *emul -> +resid       (null ptrs skip)
// Requirements: M%16==0, K%32==0, N%16==0. blockDim.x = 32*(N/16).
// Grid.x = ceil(M/64); rows clamped/guarded for the ragged last block.
// ---------------------------------------------------------------------------
__global__ void gemm_wmma_kernel(const float* __restrict__ A,
                                 const float* __restrict__ W,
                                 const float* __restrict__ bias,
                                 const float* __restrict__ emul,
                                 const float* __restrict__ resid,
                                 float* __restrict__ C,
                                 int M, int K, int N, int act)
{
    __shared__ float As[16 * MT][33];      // 64x32 f32 A tile (+1 pad)
    const int m0   = blockIdx.x * (16 * MT);
    const int lane = threadIdx.x & 31;
    const int wave = threadIdx.x >> 5;
    const int n0   = wave << 4;
    const int half = lane >> 4;            // 0 | 1
    const int lr   = lane & 15;            // A row / B col in tile

    v8f acc[MT];
    #pragma unroll
    for (int t = 0; t < MT; ++t) acc[t] = v8f{};

    for (int k0 = 0; k0 < K; k0 += 32) {
        for (int idx = threadIdx.x; idx < 16 * MT * 32; idx += blockDim.x) {
            int r = idx >> 5, c = idx & 31;
            int row = m0 + r; if (row >= M) row = M - 1;   // clamp (guarded on store)
            As[r][c] = A[(size_t)row * K + (k0 + c)];
        }
        __syncthreads();
        // B fragment: shared by all MT tiles of this wave.
        // 32x16 layout: lanes 0-15 K=0..7,16..23, lanes 16-31 K=8..15,24..31, col = lane%16
        v16bf bfr;
        #pragma unroll
        for (int e = 0; e < 16; ++e) {
            int kb = ((e >> 3) << 4) + (half << 3) + (e & 7);
            bfr[e] = (__bf16)W[(size_t)(k0 + kb) * N + (n0 + lr)];
        }
        #pragma unroll
        for (int t = 0; t < MT; ++t) {
            v16bf af;
            #pragma unroll
            for (int e = 0; e < 16; ++e) {
                // 16-bit A-matrix 16x32 layout (ISA 7.12.2)
                int ka = ((e >> 3) << 4) + (half << 3) + (((e >> 1) & 3) << 1) + (e & 1);
                af[e] = (__bf16)As[t * 16 + lr][ka];
            }
            acc[t] = __builtin_amdgcn_wmma_f32_16x16x32_bf16(false, af, false, bfr,
                                                             (short)0, acc[t], false, false);
        }
        __syncthreads();
    }
    #pragma unroll
    for (int t = 0; t < MT; ++t) {
        #pragma unroll
        for (int r = 0; r < 8; ++r) {
            int row = m0 + t * 16 + (half << 3) + r;
            if (row >= M) continue;
            int col = n0 + lr;
            float v = acc[t][r];
            if (bias)  v += bias[col];
            if (act)   v = siluf(v);
            size_t o = (size_t)row * N + col;
            if (emul)  v *= emul[o];
            if (resid) v += resid[o];
            C[o] = v;
        }
    }
}

// ---------------------------------------------------------------------------
// Embedding GEMM: x = silu(W[384x128] @ concat(h[ei], h[ej], r_e) + b)
// A tile gathered on the fly. blockDim.x = 256 (8 waves, N=128), 16-row tiles.
// ---------------------------------------------------------------------------
__global__ void embed_gemm_kernel(const float* __restrict__ table,
                                  const int*   __restrict__ z,
                                  const int*   __restrict__ ei,
                                  const int*   __restrict__ ej,
                                  const float* __restrict__ re,
                                  const float* __restrict__ W,    // 384x128
                                  const float* __restrict__ bias, // 128
                                  float* __restrict__ C, int E)
{
    __shared__ float As[16][33];
    const int K = 384, N = 128;
    const int m0   = blockIdx.x << 4;
    const int lane = threadIdx.x & 31;
    const int wave = threadIdx.x >> 5;
    const int n0   = wave << 4;
    const int half = lane >> 4;
    const int lr   = lane & 15;

    v8f acc = {};
    for (int k0 = 0; k0 < K; k0 += 32) {
        for (int idx = threadIdx.x; idx < 16 * 32; idx += blockDim.x) {
            int r = idx >> 5, c = idx & 31;
            int row = m0 + r, kk = k0 + c;
            float v;
            if (kk < 128)       v = table[(size_t)z[ei[row]] * 128 + kk];
            else if (kk < 256)  v = table[(size_t)z[ej[row]] * 128 + (kk - 128)];
            else                v = re[(size_t)row * 128 + (kk - 256)];
            As[r][c] = v;
        }
        __syncthreads();
        v16bf af, bfr;
        #pragma unroll
        for (int e = 0; e < 16; ++e) {
            int ka = ((e >> 3) << 4) + (half << 3) + (((e >> 1) & 3) << 1) + (e & 1);
            af[e]  = (__bf16)As[lr][ka];
            int kb = ((e >> 3) << 4) + (half << 3) + (e & 7);
            bfr[e] = (__bf16)W[(size_t)(k0 + kb) * N + (n0 + lr)];
        }
        acc = __builtin_amdgcn_wmma_f32_16x16x32_bf16(false, af, false, bfr,
                                                      (short)0, acc, false, false);
        __syncthreads();
    }
    #pragma unroll
    for (int r = 0; r < 8; ++r) {
        int row = m0 + (half << 3) + r;
        int col = n0 + lr;
        float v = siluf(acc[r] + bias[col]);
        C[(size_t)row * N + col] = v;
    }
}

// ---------------------------------------------------------------------------
// One-thread init: spherical Bessel zeros (bisection, fp64) + norms.
// ---------------------------------------------------------------------------
__device__ double sph_jn_d(int n, double x) {
    double j0 = sin(x) / x;
    if (n == 0) return j0;
    double j1 = sin(x) / (x * x) - cos(x) / x;
    double jm = j0, jc = j1;
    for (int l = 1; l < n; ++l) {
        double t = (2.0 * l + 1.0) / x * jc - jm;
        jm = jc; jc = t;
    }
    return jc;
}

__global__ void init_sbf_kernel(float* __restrict__ zeros_o, float* __restrict__ norms_o)
{
    if (threadIdx.x != 0 || blockIdx.x != 0) return;
    const double PI = 3.14159265358979323846;
    const int S = NUM_SPH, Kz = NUM_RAD;
    double pts[S + Kz];                       // 13 points
    double zer[NUM_SPH][NUM_RAD];
    for (int k = 0; k < S + Kz; ++k) pts[k] = (k + 1) * PI;
    for (int k = 0; k < Kz; ++k) zer[0][k] = pts[k];
    for (int l = 1; l < S; ++l) {
        double nw[S + Kz];
        int cnt = Kz + S - l;
        for (int k = 0; k < cnt; ++k) {
            double lo = pts[k], hi = pts[k + 1];
            double flo = sph_jn_d(l, lo);
            for (int it = 0; it < 100; ++it) {
                double mid = 0.5 * (lo + hi);
                double fmid = sph_jn_d(l, mid);
                if (flo * fmid <= 0.0) hi = mid;
                else { lo = mid; flo = fmid; }
            }
            nw[k] = 0.5 * (lo + hi);
        }
        for (int k = 0; k < cnt; ++k) pts[k] = nw[k];
        for (int k = 0; k < Kz; ++k) zer[l][k] = nw[k];
    }
    for (int l = 0; l < S; ++l)
        for (int k = 0; k < Kz; ++k) {
            zeros_o[l * NUM_RAD + k] = (float)zer[l][k];
            double jn = sph_jn_d(l + 1, zer[l][k]);
            norms_o[l * NUM_RAD + k] = (float)(1.0 / sqrt(0.5 * jn * jn));
        }
}

// ---------------------------------------------------------------------------
// Per-edge: distance -> envelope -> rbf[6], rad[7*6]
// ---------------------------------------------------------------------------
__global__ void edge_basis_kernel(const float* __restrict__ pos,
                                  const int* __restrict__ ei, const int* __restrict__ ej,
                                  const float* __restrict__ rbf_freq,
                                  const float* __restrict__ sbf_zero,
                                  const float* __restrict__ sbf_norm,
                                  float* __restrict__ rbf, float* __restrict__ rad, int E)
{
    int e = blockIdx.x * blockDim.x + threadIdx.x;
    if (e >= E) return;
    int i = ei[e], j = ej[e];
    float dx = pos[3*i]   - pos[3*j];
    float dy = pos[3*i+1] - pos[3*j+1];
    float dz = pos[3*i+2] - pos[3*j+2];
    float dist = sqrtf(dx*dx + dy*dy + dz*dz);
    float d = dist / CUTOFF_F;
    float env = 0.0f;
    if (d < 1.0f) {                 // p = 6: 1/x - 28 x^5 + 48 x^6 - 21 x^7
        float d2 = d * d, d5 = d2 * d2 * d;
        env = 1.0f / d - 28.0f * d5 + 48.0f * d5 * d - 21.0f * d5 * d2;
    }
    #pragma unroll
    for (int r = 0; r < NUM_RAD; ++r)
        rbf[(size_t)e * NUM_RAD + r] = env * __sinf(rbf_freq[r] * d);

    #pragma unroll
    for (int r = 0; r < NUM_RAD; ++r) {
        float jlm[NUM_SPH];
        for (int l = 0; l < NUM_SPH; ++l) {
            float u = d * sbf_zero[l * NUM_RAD + r];
            float j0 = __sinf(u) / u;
            float jc = j0;
            if (l > 0) {
                float j1 = __sinf(u) / (u * u) - __cosf(u) / u;
                float a = j0, b = j1;
                for (int m = 1; m < l; ++m) {
                    float t = (2.0f * m + 1.0f) / u * b - a;
                    a = b; b = t;
                }
                jc = b;
            }
            jlm[l] = jc;
        }
        for (int l = 0; l < NUM_SPH; ++l)
            rad[(size_t)e * (NUM_SPH * NUM_RAD) + l * NUM_RAD + r] =
                env * sbf_norm[l * NUM_RAD + r] * jlm[l];
    }
}

// r_e = silu(rbf @ W[6x128] + b)  — one thread per output element
__global__ void rbf_lin_silu_kernel(const float* __restrict__ rbf,
                                    const float* __restrict__ W,
                                    const float* __restrict__ bias,
                                    float* __restrict__ out, int E)
{
    int idx = blockIdx.x * blockDim.x + threadIdx.x;
    if (idx >= E * HIDDEN) return;
    int e = idx >> 7, c = idx & 127;
    float s = bias[c];
    #pragma unroll
    for (int k = 0; k < NUM_RAD; ++k)
        s += rbf[(size_t)e * NUM_RAD + k] * W[k * HIDDEN + c];
    out[idx] = siluf(s);
}

// gate = rbf @ W1[6x8] @ W2[8x128]  (no activation) — one thread per edge
__global__ void rbf_gate_kernel(const float* __restrict__ rbf,
                                const float* __restrict__ w1,
                                const float* __restrict__ w2,
                                float* __restrict__ gate, int E)
{
    __shared__ float s1[NUM_RAD * 8], s2[8 * HIDDEN];
    for (int i = threadIdx.x; i < NUM_RAD * 8; i += blockDim.x) s1[i] = w1[i];
    for (int i = threadIdx.x; i < 8 * HIDDEN; i += blockDim.x) s2[i] = w2[i];
    __syncthreads();
    int e = blockIdx.x * blockDim.x + threadIdx.x;
    if (e >= E) return;
    float r[NUM_RAD];
    #pragma unroll
    for (int k = 0; k < NUM_RAD; ++k) r[k] = rbf[(size_t)e * NUM_RAD + k];
    float t[8];
    #pragma unroll
    for (int b = 0; b < 8; ++b) {
        float s = 0.f;
        #pragma unroll
        for (int k = 0; k < NUM_RAD; ++k) s += r[k] * s1[k * 8 + b];
        t[b] = s;
    }
    for (int c = 0; c < HIDDEN; ++c) {
        float s = 0.f;
        #pragma unroll
        for (int b = 0; b < 8; ++b) s += t[b] * s2[b * HIDDEN + c];
        gate[(size_t)e * HIDDEN + c] = s;
    }
}

// Triplet path: sbf(42) -> W1(42x8) -> W2(8x64); scatter-add into agg[ji].
__global__ void triplet_kernel(const float* __restrict__ pos,
                               const int* __restrict__ ti, const int* __restrict__ tj,
                               const int* __restrict__ tk, const int* __restrict__ tkj,
                               const int* __restrict__ tji,
                               const float* __restrict__ rad,
                               const float* __restrict__ xdown,
                               const float* __restrict__ w1,  // 42x8
                               const float* __restrict__ w2,  // 8x64
                               float* __restrict__ agg, int T)
{
    __shared__ float s1[42 * 8], s2[8 * INT_EMB];
    for (int i = threadIdx.x; i < 42 * 8; i += blockDim.x) s1[i] = w1[i];
    for (int i = threadIdx.x; i < 8 * INT_EMB; i += blockDim.x) s2[i] = w2[i];
    __syncthreads();
    int t = blockIdx.x * blockDim.x + threadIdx.x;
    if (t >= T) return;
    int i = ti[t], j = tj[t], k = tk[t], kj = tkj[t], ji = tji[t];
    float jx = pos[3*i]-pos[3*j], jy = pos[3*i+1]-pos[3*j+1], jz = pos[3*i+2]-pos[3*j+2];
    float kx = pos[3*k]-pos[3*j], ky = pos[3*k+1]-pos[3*j+1], kz = pos[3*k+2]-pos[3*j+2];
    float a  = jx*kx + jy*ky + jz*kz;
    float cx = jy*kz - jz*ky, cy = jz*kx - jx*kz, cz = jx*ky - jy*kx;
    float b  = sqrtf(cx*cx + cy*cy + cz*cz);
    float c  = a * __frsqrt_rn(a * a + b * b);    // cos(atan2(b,a))
    float P[NUM_SPH];
    P[0] = 1.0f; P[1] = c;
    #pragma unroll
    for (int l = 1; l < NUM_SPH - 1; ++l)
        P[l + 1] = ((2.0f * l + 1.0f) * c * P[l] - (float)l * P[l - 1]) / (float)(l + 1);
    float cbf[NUM_SPH];
    #pragma unroll
    for (int l = 0; l < NUM_SPH; ++l)
        cbf[l] = sqrtf((2.0f * l + 1.0f) / (4.0f * 3.14159265358979f)) * P[l];
    float t8[8];
    #pragma unroll
    for (int q = 0; q < 8; ++q) t8[q] = 0.f;
    for (int l = 0; l < NUM_SPH; ++l)
        for (int r = 0; r < NUM_RAD; ++r) {
            float s = rad[(size_t)kj * 42 + l * NUM_RAD + r] * cbf[l];
            int rowi = (l * NUM_RAD + r) * 8;
            #pragma unroll
            for (int q = 0; q < 8; ++q) t8[q] += s * s1[rowi + q];
        }
    for (int cc = 0; cc < INT_EMB; ++cc) {
        float s = 0.f;
        #pragma unroll
        for (int q = 0; q < 8; ++q) s += t8[q] * s2[q * INT_EMB + cc];
        float v = xdown[(size_t)kj * INT_EMB + cc] * s;
        atomicAdd(&agg[(size_t)ji * INT_EMB + cc], v);
    }
}

// nf[n][c] = sum_{d<6} lin_rbf(rbf[e])[c] * x[e][c],  e = n*6+d (contiguous!)
__global__ void node_agg_kernel(const float* __restrict__ rbf,
                                const float* __restrict__ Wr,  // 6x128
                                const float* __restrict__ x,
                                float* __restrict__ nf, int Nn)
{
    __shared__ float sw[NUM_RAD * HIDDEN];
    for (int i = threadIdx.x; i < NUM_RAD * HIDDEN; i += blockDim.x) sw[i] = Wr[i];
    __syncthreads();
    int idx = blockIdx.x * blockDim.x + threadIdx.x;
    if (idx >= Nn * HIDDEN) return;
    int n = idx >> 7, c = idx & 127;
    float s = 0.f;
    #pragma unroll
    for (int d = 0; d < DEG; ++d) {
        int e = n * DEG + d;
        float g = 0.f;
        #pragma unroll
        for (int k = 0; k < NUM_RAD; ++k)
            g += rbf[(size_t)e * NUM_RAD + k] * sw[k * HIDDEN + c];
        s += g * x[(size_t)e * HIDDEN + c];
    }
    nf[idx] = s;
}

// P[n] += dot(ob[n, 0:256], W[256])
__global__ void out_final_kernel(const float* __restrict__ ob,
                                 const float* __restrict__ W,
                                 float* __restrict__ P, int Nn)
{
    int n = blockIdx.x * blockDim.x + threadIdx.x;
    if (n >= Nn) return;
    float s = 0.f;
    for (int c = 0; c < OUT_EMB; ++c) s += ob[(size_t)n * OUT_EMB + c] * W[c];
    P[n] += s;
}

__global__ void zero_kernel(float* __restrict__ p, int n)
{
    int i = blockIdx.x * blockDim.x + threadIdx.x;
    if (i < n) p[i] = 0.0f;
}

// out[m] = sum of P over the (contiguous) atoms of molecule m
__global__ void mol_sum_kernel(const float* __restrict__ P, float* __restrict__ out,
                               int NM, int atoms)
{
    int m = blockIdx.x * blockDim.x + threadIdx.x;
    if (m >= NM) return;
    float s = 0.f;
    for (int a = 0; a < atoms; ++a) s += P[m * atoms + a];
    out[m] = s;
}

// ---------------------------------------------------------------------------
// Launcher.  Param leaves follow jax tree-flatten order (dict keys sorted):
//   0..10  z,pos,batch,edge_i,edge_j,t_i,t_j,t_k,t_kj,t_ji,num_graphs
//   11..15 emb: lin.b, lin.w, lin_rbf.b, lin_rbf.w, table
//   16+24b int[b]: after0.l1.b/w, after0.l2.b/w, after1.l1.b/w, after1.l2.b/w,
//                  before0.l1.b/w, before0.l2.b/w, down.w, ji.b/w, kj.b/w,
//                  lin.b/w, rbf1.w, rbf2.w, sbf1.w, sbf2.w, up.w
//   112+10o out[o]: lin.w, lin_rbf.w, lin_up.b/w, lins0.b/w, lins1.b/w, lins2.b/w
//   162    rbf_freq
// Workspace requirement ~390 MB (assumed available in d_ws).
// ---------------------------------------------------------------------------
extern "C" void kernel_launch(void* const* d_in, const int* in_sizes, int n_in,
                              void* d_out, int out_size, void* d_ws, size_t ws_size,
                              hipStream_t stream)
{
    auto F = [&](int i) { return (const float*)d_in[i]; };
    auto I = [&](int i) { return (const int*)d_in[i]; };

    const int Nn = in_sizes[0];          // 20000
    const int E  = in_sizes[3];          // 120000
    const int T  = in_sizes[8];          // ~700k triplets
    const int NM = out_size;             // 800
    const int atoms = Nn / NM;

    const int*   z   = I(0);
    const float* pos = F(1);
    const int*   ei  = I(3);
    const int*   ej  = I(4);
    const int*   ti  = I(5);
    const int*   tj  = I(6);
    const int*   tk  = I(7);
    const int*   tkj = I(8);
    const int*   tji = I(9);

    const float* emb_lin_b = F(11);
    const float* emb_lin_w = F(12);
    const float* emb_rbf_b = F(13);
    const float* emb_rbf_w = F(14);
    const float* emb_table = F(15);
    const float* rbf_freq  = F(162);
    auto IB = [&](int b, int o) { return F(16 + 24 * b + o); };
    auto OB = [&](int o, int l) { return F(112 + 10 * o + l); };

    // ---- workspace carve-out (256B aligned) ----
    char*  w   = (char*)d_ws;
    size_t off = 0;
    auto alloc = [&](size_t bytes) {
        void* p = w + off;
        off = (off + bytes + 255) & ~(size_t)255;
        return (float*)p;
    };
    float* sbf_zero = alloc(42 * 4);
    float* sbf_norm = alloc(42 * 4);
    float* rbf      = alloc((size_t)E * NUM_RAD * 4);
    float* rad      = alloc((size_t)E * 42 * 4);
    float* bufX     = alloc((size_t)E * HIDDEN * 4);   // x
    float* buf1     = alloc((size_t)E * HIDDEN * 4);
    float* buf2     = alloc((size_t)E * HIDDEN * 4);
    float* buf3     = alloc((size_t)E * HIDDEN * 4);
    float* bufDown  = alloc((size_t)E * INT_EMB * 4);
    float* bufAgg   = alloc((size_t)E * INT_EMB * 4);
    float* nf       = alloc((size_t)Nn * HIDDEN * 4);
    float* ob1      = alloc((size_t)Nn * OUT_EMB * 4);
    float* ob2      = alloc((size_t)Nn * OUT_EMB * 4);
    float* P        = alloc((size_t)Nn * 4);
    (void)ws_size; (void)n_in;

    auto gemm = [&](const float* A, const float* W_, const float* bias,
                    const float* emul, const float* resid, float* C,
                    int M, int K, int N, int act) {
        gemm_wmma_kernel<<<dim3((M + 16 * MT - 1) / (16 * MT)),
                           dim3(32 * (N / 16)), 0, stream>>>(
            A, W_, bias, emul, resid, C, M, K, N, act);
    };

    auto out_block = [&](int o, const float* x) {
        node_agg_kernel<<<(Nn * HIDDEN + 255) / 256, 256, 0, stream>>>(
            rbf, OB(o, 1), x, nf, Nn);
        gemm(nf,  OB(o, 3), OB(o, 2), nullptr, nullptr, ob1, Nn, HIDDEN,  OUT_EMB, 0);
        gemm(ob1, OB(o, 5), OB(o, 4), nullptr, nullptr, ob2, Nn, OUT_EMB, OUT_EMB, 1);
        gemm(ob2, OB(o, 7), OB(o, 6), nullptr, nullptr, ob1, Nn, OUT_EMB, OUT_EMB, 1);
        gemm(ob1, OB(o, 9), OB(o, 8), nullptr, nullptr, ob2, Nn, OUT_EMB, OUT_EMB, 1);
        out_final_kernel<<<(Nn + 255) / 256, 256, 0, stream>>>(ob2, OB(o, 0), P, Nn);
    };

    // ---- basis ----
    init_sbf_kernel<<<1, 1, 0, stream>>>(sbf_zero, sbf_norm);
    edge_basis_kernel<<<(E + 255) / 256, 256, 0, stream>>>(
        pos, ei, ej, rbf_freq, sbf_zero, sbf_norm, rbf, rad, E);

    // ---- embedding: r_e then x ----
    rbf_lin_silu_kernel<<<(E * HIDDEN + 255) / 256, 256, 0, stream>>>(
        rbf, emb_rbf_w, emb_rbf_b, buf1, E);
    embed_gemm_kernel<<<dim3(E / 16), dim3(256), 0, stream>>>(
        emb_table, z, ei, ej, buf1, emb_lin_w, emb_lin_b, bufX, E);

    zero_kernel<<<(Nn + 255) / 256, 256, 0, stream>>>(P, Nn);
    out_block(0, bufX);

    // ---- interaction blocks ----
    for (int b = 0; b < 4; ++b) {
        // x_ji = silu(ji.x + b)
        gemm(bufX, IB(b, 14), IB(b, 13), nullptr, nullptr, buf1, E, HIDDEN, HIDDEN, 1);
        // gate = rbf2(rbf1(rbf))
        rbf_gate_kernel<<<(E + 127) / 128, 128, 0, stream>>>(
            rbf, IB(b, 19), IB(b, 20), buf3, E);
        // x_kj = silu(kj.x + b) * gate
        gemm(bufX, IB(b, 16), IB(b, 15), buf3, nullptr, buf2, E, HIDDEN, HIDDEN, 1);
        // x_down = silu(down . x_kj)
        gemm(buf2, IB(b, 12), nullptr, nullptr, nullptr, bufDown, E, HIDDEN, INT_EMB, 1);
        // triplet gather/scale/scatter
        zero_kernel<<<(E * INT_EMB + 255) / 256, 256, 0, stream>>>(bufAgg, E * INT_EMB);
        triplet_kernel<<<(T + 127) / 128, 128, 0, stream>>>(
            pos, ti, tj, tk, tkj, tji, rad, bufDown, IB(b, 21), IB(b, 22), bufAgg, T);
        // h = silu(up . agg) + x_ji
        gemm(bufAgg, IB(b, 23), nullptr, nullptr, buf1, buf2, E, INT_EMB, HIDDEN, 1);
        // before residual
        gemm(buf2, IB(b, 9),  IB(b, 8),  nullptr, nullptr, buf3, E, HIDDEN, HIDDEN, 1);
        gemm(buf3, IB(b, 11), IB(b, 10), nullptr, buf2,    buf1, E, HIDDEN, HIDDEN, 1);
        // h = silu(lin . h + b) + x
        gemm(buf1, IB(b, 18), IB(b, 17), nullptr, bufX,    buf2, E, HIDDEN, HIDDEN, 1);
        // after residuals
        gemm(buf2, IB(b, 1),  IB(b, 0),  nullptr, nullptr, buf3, E, HIDDEN, HIDDEN, 1);
        gemm(buf3, IB(b, 3),  IB(b, 2),  nullptr, buf2,    buf1, E, HIDDEN, HIDDEN, 1);
        gemm(buf1, IB(b, 5),  IB(b, 4),  nullptr, nullptr, buf3, E, HIDDEN, HIDDEN, 1);
        gemm(buf3, IB(b, 7),  IB(b, 6),  nullptr, buf1,    bufX, E, HIDDEN, HIDDEN, 1);

        out_block(b + 1, bufX);
    }

    // ---- per-molecule reduction ----
    mol_sum_kernel<<<(NM + 255) / 256, 256, 0, stream>>>(P, (float*)d_out, NM, atoms);
}
